// Graphormer_7842610282923
// MI455X (gfx1250) — matrix operand order repaired and checked by reading, not compiled, not run on hardware
//
#include <hip/hip_runtime.h>
#include <hip/hip_bf16.h>
#include <stdint.h>

// ---------------- constants ----------------
constexpr int DM     = 512;          // model dim
constexpr int Bb     = 16;           // batch
constexpr int Ss     = 512;          // seq len
constexpr int Hh     = 8;            // heads
constexpr int HDim   = 64;           // head dim
constexpr int Ll     = 6;            // layers
constexpr int Nnodes = Bb * (Ss - 1);

typedef _Float16 half_t;
typedef _Float16 v16h  __attribute__((ext_vector_type(16)));
typedef float    v8f   __attribute__((ext_vector_type(8)));
typedef unsigned int u32x4 __attribute__((ext_vector_type(4)));
typedef int      i32x4 __attribute__((ext_vector_type(4)));
typedef int      i32x8 __attribute__((ext_vector_type(8)));

union U32H2 { uint32_t u; _Float16 h[2]; uint16_t us[2]; };

#define HAVE_TDM __has_builtin(__builtin_amdgcn_tensor_load_to_lds)

__device__ __forceinline__ v8f wmma16(v16h a, v16h b, v8f c) {
    // D = A(16x32) * B(32x16) + C, f16 in / f32 acc -> v_wmma_f32_16x16x32_f16
    return __builtin_amdgcn_wmma_f32_16x16x32_f16(false, a, false, b, (short)0, c,
                                                  false, false);
}

// A operand (16x32, f16 source, row-major, stride in halves).
// Layout (ISA 7.12.2): lanes 0-15 / 16-31 hold M=0..15; pair v<4 -> K=8g+2v,
// v>=4 -> K=16+8g+2(v-4).
__device__ __forceinline__ v16h load_a_f16(const half_t* base, int stride) {
    int lane = threadIdx.x & 31;
    int g = lane >> 4, m = lane & 15;
    v16h a;
#pragma unroll
    for (int vv = 0; vv < 8; ++vv) {
        int kk = (vv < 4) ? (8 * g + 2 * vv) : (16 + 8 * g + 2 * (vv - 4));
        U32H2 cv;
        cv.u = *(const uint32_t*)(base + (size_t)m * stride + kk);
        a[2 * vv]     = cv.h[0];
        a[2 * vv + 1] = cv.h[1];
    }
    return a;
}

// Same A layout but f32 source (converted to f16 on load).
__device__ __forceinline__ v16h load_a_f32(const float* base, int stride) {
    int lane = threadIdx.x & 31;
    int g = lane >> 4, m = lane & 15;
    v16h a;
#pragma unroll
    for (int vv = 0; vv < 8; ++vv) {
        int kk = (vv < 4) ? (8 * g + 2 * vv) : (16 + 8 * g + 2 * (vv - 4));
        a[2 * vv]     = (half_t)base[m * stride + kk];
        a[2 * vv + 1] = (half_t)base[m * stride + kk + 1];
    }
    return a;
}

// B operand (32x16, K x N) read from a TRANSPOSED source laid out [N][K]
// (stride in halves). Same K layout as A: pair v<4 -> K=8g+2v,
// v>=4 -> K=16+8g+2(v-4); N = lane&15. K-pairs are contiguous 32-bit loads.
__device__ __forceinline__ v16h load_bt_f16(const half_t* baseT, int stride) {
    int lane = threadIdx.x & 31;
    int g = lane >> 4, n = lane & 15;
    v16h b;
#pragma unroll
    for (int vv = 0; vv < 8; ++vv) {
        int kk = (vv < 4) ? (8 * g + 2 * vv) : (16 + 8 * g + 2 * (vv - 4));
        U32H2 cv;
        cv.u = *(const uint32_t*)(baseT + (size_t)n * stride + kk);
        b[2 * vv]     = cv.h[0];
        b[2 * vv + 1] = cv.h[1];
    }
    return b;
}

#if HAVE_TDM
// Issue a TDM 2-D tile load: tile_rows x 32 halves from a [tensor_rows x 512]
// f16 tensor into LDS with a 40-half padded row stride (pad 4 DWORDs every
// 16 DWORDs), matching the WMMA operand readers. Wave-level op; caller must
// s_wait_tensorcnt then barrier.
__device__ __forceinline__ void tdm_load_tile_2d(uint32_t lds_byte_addr,
                                                 const half_t* gaddr,
                                                 uint32_t tensor_rows,
                                                 uint32_t tile_rows) {
    uint64_t ga = (uint64_t)(uintptr_t)gaddr;
    u32x4 g0;
    g0[0] = 1u;                                   // count=1, user descriptor
    g0[1] = lds_byte_addr;                        // LDS dest (bytes)
    g0[2] = (uint32_t)ga;                         // global_addr[31:0]
    g0[3] = (uint32_t)((ga >> 32) & 0x01FFFFFFu)  // global_addr[56:32]
            | 0x80000000u;                        // type=2 ("image")
    i32x8 g1;
    g1[0] = (int)((1u << 16)      // data_size = 2 bytes
                | (1u << 20)      // pad_enable
                | (3u << 22)      // pad_interval: 16 DWORDs
                | (3u << 25));    // pad_amount:   4 DWORDs  -> stride 40 halves
    g1[1] = (int)((uint32_t)DM << 16);            // tensor_dim0 = 512 (lo16)
    g1[2] = (int)((tensor_rows & 0xFFFFu) << 16); // tensor_dim1 lo16
    g1[3] = (int)(((tensor_rows >> 16) & 0xFFFFu) // tensor_dim1 hi16
                | (32u << 16));                   // tile_dim0 = 32
    g1[4] = (int)tile_rows;                       // tile_dim1
    g1[5] = (int)DM;                              // tensor_dim0_stride = 512
    g1[6] = 0;
    g1[7] = 0;
    i32x4 z4 = {0, 0, 0, 0};
#if __clang_major__ >= 23
    i32x8 z8 = {0, 0, 0, 0, 0, 0, 0, 0};
    __builtin_amdgcn_tensor_load_to_lds(g0, g1, z4, z4, z8, 0);
#else
    __builtin_amdgcn_tensor_load_to_lds(g0, g1, z4, z4, 0);
#endif
}
#endif

// ---------------- weight convert + transpose (f32 [K][N] -> f16 [N][K]) -----
__global__ __launch_bounds__(256) void cvt_weights_kernel(
    const float* __restrict__ Wq, const float* __restrict__ Wk,
    const float* __restrict__ Wv, const float* __restrict__ Wo,
    const float* __restrict__ W1, const float* __restrict__ W2,
    half_t* __restrict__ dst) {
    int z = blockIdx.z;            // 0..35
    int l = z / 6, t = z % 6;
    const float* src = (t == 0 ? Wq : t == 1 ? Wk : t == 2 ? Wv
                       : t == 3 ? Wo : t == 4 ? W1 : W2) +
                       (size_t)l * DM * DM;
    half_t* d = dst + (size_t)z * DM * DM;
    int idx = blockIdx.x * 256 + threadIdx.x;    // over K*N, grid.x = 1024
    int k = idx >> 9, n = idx & 511;
    d[(size_t)n * DM + k] = (half_t)src[idx];
}

// ---------------- node embedding:  h = x@node_W + node_b + deg_tab[deg] ----
__global__ __launch_bounds__(256) void embed_kernel(
    const float* __restrict__ x, const int* __restrict__ deg,
    const float* __restrict__ nodeW, const float* __restrict__ nodeb,
    const float* __restrict__ degtab, const float* __restrict__ vnode,
    float* __restrict__ h) {
    int r = blockIdx.y;
    int d = blockIdx.x * 256 + threadIdx.x;
    __shared__ float xr[64];
    if (r < Nnodes) {
        if (threadIdx.x < 64) xr[threadIdx.x] = x[(size_t)r * 64 + threadIdx.x];
        __syncthreads();
        float acc = nodeb[d] + degtab[(size_t)deg[r] * DM + d];
#pragma unroll 8
        for (int k = 0; k < 64; ++k) acc += xr[k] * nodeW[(size_t)k * DM + d];
        h[(size_t)r * DM + d] = acc;
    } else {
        h[(size_t)r * DM + d] = vnode[d];
    }
}

// ---------------- w_comb[e] = sum_d edge_weight[e,d]*edge_proj_W[d] --------
__global__ void wcomb_kernel(const float* __restrict__ ew,
                             const float* __restrict__ epW,
                             float* __restrict__ wc) {
    int e = threadIdx.x;
    if (e < 16) {
        float s = 0.f;
        for (int d = 0; d < DM; ++d) s += ew[e * DM + d] * epW[d];
        wc[e] = s;
    }
}

// ---------------- fused attention bias: sp + edge + mask -------------------
__global__ __launch_bounds__(256) void bias_kernel(
    const int* __restrict__ sp, const float* __restrict__ sptab,
    const float* __restrict__ ea, const float* __restrict__ wc,
    const float* __restrict__ epb, const int* __restrict__ lengths,
    float* __restrict__ bias) {
    __shared__ float wcs[16];
    if (threadIdx.x < 16) wcs[threadIdx.x] = wc[threadIdx.x];
    __syncthreads();
    size_t idx = (size_t)blockIdx.x * 256 + threadIdx.x;
    int j = (int)(idx & 511);
    int b = (int)(idx >> 18);                   // 512*512 = 2^18
    float v = sptab[sp[idx] + 1] + epb[0];
    const float* e = ea + idx * 16;
#pragma unroll
    for (int t = 0; t < 16; ++t) v += e[t] * wcs[t];
    if (j > lengths[b]) v = -1e30f;             // key-padding mask folded in
    bias[idx] = v;
}

// ---------------- LayerNorm (row = block), writes f16 ----------------------
__global__ __launch_bounds__(256) void ln_kernel(
    const float* __restrict__ h, const float* __restrict__ sc,
    const float* __restrict__ bi, half_t* __restrict__ out) {
    int row = blockIdx.x, tid = threadIdx.x;
    const float* x = h + (size_t)row * DM;
    float x0 = x[tid], x1 = x[tid + 256];
    float sum = x0 + x1, sq = x0 * x0 + x1 * x1;
#pragma unroll
    for (int off = 16; off; off >>= 1) {
        sum += __shfl_xor(sum, off, 32);
        sq  += __shfl_xor(sq,  off, 32);
    }
    __shared__ float ssum[8], ssq[8];
    int wv = tid >> 5, ln = tid & 31;
    if (ln == 0) { ssum[wv] = sum; ssq[wv] = sq; }
    __syncthreads();
    float ts = 0.f, tq = 0.f;
#pragma unroll
    for (int i = 0; i < 8; ++i) { ts += ssum[i]; tq += ssq[i]; }
    float mean = ts * (1.f / DM);
    float var  = tq * (1.f / DM) - mean * mean;
    float inv  = rsqrtf(var + 1e-5f);
    out[(size_t)row * DM + tid]       = (half_t)((x0 - mean) * inv * sc[tid] + bi[tid]);
    out[(size_t)row * DM + tid + 256] = (half_t)((x1 - mean) * inv * sc[tid + 256] + bi[tid + 256]);
}

// ---------------- generic WMMA GEMM: C[M,512] = A[M,512]@W + b -------------
// A f16 row-major; Wt f16 transposed [N][K]; optional relu / residual / f32 /
// f16 outputs. Tile: 128x64 per block (8 waves, each 32x32 = 4 WMMA tiles).
// A/W tiles staged into LDS by the Tensor Data Mover when available.
template <int RELU, int RESID, int OUTF32, int OUTF16>
__global__ __launch_bounds__(256) void gemm_kernel(
    const half_t* __restrict__ A, const half_t* __restrict__ Wt,
    const float* __restrict__ bias, const float* __restrict__ res,
    float* __restrict__ outf, half_t* __restrict__ outh) {
    __shared__ half_t As[128 * 40];   // stride 40 halves (16B-aligned rows)
    __shared__ half_t Ws[64 * 40];
    int tid = threadIdx.x;
    int wave = tid >> 5;
    int wm = wave & 3, wn = wave >> 2;        // wave grid 4(M) x 2(N)
    int mBase = blockIdx.x * 128;
    int nBase = blockIdx.y * 64;

    v8f zero = {0.f, 0.f, 0.f, 0.f, 0.f, 0.f, 0.f, 0.f};
    v8f acc[2][2] = {{zero, zero}, {zero, zero}};

    for (int it = 0; it < DM / 32; ++it) {
        int k0 = it * 32;
#if HAVE_TDM
        // TDM DMA: wave0 -> A tile, wave1 -> W tile
        if (tid == 0) {
            tdm_load_tile_2d((uint32_t)(uintptr_t)&As[0],
                             A + (size_t)mBase * DM + k0, Bb * Ss, 128);
        } else if (tid == 32) {
            tdm_load_tile_2d((uint32_t)(uintptr_t)&Ws[0],
                             Wt + (size_t)nBase * DM + k0, DM, 64);
        }
        if (tid < 64) __builtin_amdgcn_s_wait_tensorcnt(0);
#else
        {   // stage A tile: 128 rows x 32 halves, 16 halves / thread
            int row = tid >> 1;
            int col = (tid & 1) * 16;
            const uint4* src =
                (const uint4*)(A + (size_t)(mBase + row) * DM + k0 + col);
            uint4* dst = (uint4*)(&As[row * 40 + col]);
            dst[0] = src[0];
            dst[1] = src[1];
        }
        {   // stage W^T tile: 64 rows x 32 halves, 8 halves / thread
            int row = tid >> 2;
            int col = (tid & 3) * 8;
            *(uint4*)(&Ws[row * 40 + col]) =
                *(const uint4*)(Wt + (size_t)(nBase + row) * DM + k0 + col);
        }
        if (it + 1 < DM / 32) {   // hint next tiles -> global_prefetch
            __builtin_prefetch(A + (size_t)(mBase + (tid >> 1)) * DM + k0 + 32, 0, 1);
            __builtin_prefetch(Wt + (size_t)(nBase + (tid >> 2)) * DM + k0 + 32, 0, 1);
        }
#endif
        __syncthreads();
        v16h a0 = load_a_f16(&As[(wm * 32 + 0)  * 40], 40);
        v16h a1 = load_a_f16(&As[(wm * 32 + 16) * 40], 40);
        v16h b0 = load_bt_f16(&Ws[(wn * 32 + 0)  * 40], 40);
        v16h b1 = load_bt_f16(&Ws[(wn * 32 + 16) * 40], 40);
        acc[0][0] = wmma16(a0, b0, acc[0][0]);
        acc[0][1] = wmma16(a0, b1, acc[0][1]);
        acc[1][0] = wmma16(a1, b0, acc[1][0]);
        acc[1][1] = wmma16(a1, b1, acc[1][1]);
        __syncthreads();
    }
    int lane = tid & 31, g = lane >> 4, n = lane & 15;
#pragma unroll
    for (int sm = 0; sm < 2; ++sm)
#pragma unroll
        for (int sn = 0; sn < 2; ++sn)
#pragma unroll
            for (int r = 0; r < 8; ++r) {
                int row = mBase + wm * 32 + sm * 16 + r + 8 * g;
                int col = nBase + wn * 32 + sn * 16 + n;
                float vv = acc[sm][sn][r] + bias[col];
                if (RELU)  vv = fmaxf(vv, 0.f);
                if (RESID) vv += res[(size_t)row * DM + col];
                if (OUTF32) outf[(size_t)row * DM + col] = vv;
                if (OUTF16) outh[(size_t)row * DM + col] = (half_t)vv;
            }
}

// ---------------- attention: one wave per (b, head, 16-query tile) ---------
__global__ __launch_bounds__(32) void attn_kernel(
    const half_t* __restrict__ q, const half_t* __restrict__ k,
    const half_t* __restrict__ v, const float* __restrict__ bias,
    half_t* __restrict__ o) {
    __shared__ float  sc[16 * 512];     // scores / probabilities (32 KB)
    __shared__ half_t vt[64 * 34];      // V^T tile [hd][key], padded
    __shared__ float  sinv[16];         // per-row 1/sum (deferred softmax norm)
    int qt = blockIdx.x, hd = blockIdx.y, b = blockIdx.z;
    int lane = threadIdx.x;
    int g = lane >> 4, ln = lane & 15;
    v8f zero = {0.f, 0.f, 0.f, 0.f, 0.f, 0.f, 0.f, 0.f};

    // Q tile operands (16 rows x HD=64 -> two K=32 chunks)
    const half_t* qbase = q + ((size_t)(b * Ss + qt * 16) * DM + hd * HDim);
    v16h aq0 = load_a_f16(qbase, DM);
    v16h aq1 = load_a_f16(qbase + 32, DM);

    // scores = scale * q k^T + bias   (key tiles of 16)
    for (int kt = 0; kt < 32; ++kt) {
        const half_t* kbase = k + ((size_t)(b * Ss + kt * 16) * DM + hd * HDim);
        v16h b0 = load_bt_f16(kbase, DM);       // B[K=hd, N=key] = k[key][hd]
        v16h b1 = load_bt_f16(kbase + 32, DM);
        v8f c = zero;
        c = wmma16(aq0, b0, c);
        c = wmma16(aq1, b1, c);
#pragma unroll
        for (int r = 0; r < 8; ++r) {
            int m = r + 8 * g;                  // query row within tile
            int qg = qt * 16 + m;
            int kg = kt * 16 + ln;
            sc[m * 512 + kg] =
                c[r] * 0.125f + bias[((size_t)b * Ss + qg) * Ss + kg];
        }
    }
    __syncthreads();

    // softmax max+exp (2 lanes per row, 256 cols each, float4-vectorized);
    // normalization deferred to the output accumulators via sinv[]
    {
        int row = lane >> 1, p = lane & 1;
        float* rp = &sc[row * 512 + p * 256];
        float4* rp4 = (float4*)rp;
        float mx = -3.0e38f;
        for (int i = 0; i < 64; ++i) {
            float4 t = rp4[i];
            mx = fmaxf(mx, fmaxf(fmaxf(t.x, t.y), fmaxf(t.z, t.w)));
        }
        mx = fmaxf(mx, __shfl_xor(mx, 1, 32));
        float sum = 0.f;
        for (int i = 0; i < 64; ++i) {
            float4 t = rp4[i];
            t.x = __expf(t.x - mx);
            t.y = __expf(t.y - mx);
            t.z = __expf(t.z - mx);
            t.w = __expf(t.w - mx);
            rp4[i] = t;
            sum += (t.x + t.y) + (t.z + t.w);
        }
        sum += __shfl_xor(sum, 1, 32);
        if (p == 0) sinv[row] = 1.f / sum;
    }
    __syncthreads();

    // out = P @ V, streaming 32 keys per step, 4 hd tiles of 16
    v8f co[4] = {zero, zero, zero, zero};
    for (int s = 0; s < 16; ++s) {
        {   // stage V^T: lane loads key row s*32+lane (64 contiguous halves)
            const half_t* vrow =
                v + ((size_t)(b * Ss + s * 32 + lane) * DM + hd * HDim);
#pragma unroll
            for (int i = 0; i < 8; ++i) {
                uint4 dd = ((const uint4*)vrow)[i];
                U32H2 c0, c1, c2, c3;
                c0.u = dd.x; c1.u = dd.y; c2.u = dd.z; c3.u = dd.w;
                int hb = i * 8;
                vt[(hb + 0) * 34 + lane] = c0.h[0];
                vt[(hb + 1) * 34 + lane] = c0.h[1];
                vt[(hb + 2) * 34 + lane] = c1.h[0];
                vt[(hb + 3) * 34 + lane] = c1.h[1];
                vt[(hb + 4) * 34 + lane] = c2.h[0];
                vt[(hb + 5) * 34 + lane] = c2.h[1];
                vt[(hb + 6) * 34 + lane] = c3.h[0];
                vt[(hb + 7) * 34 + lane] = c3.h[1];
            }
        }
        __syncthreads();
        v16h ap = load_a_f32(&sc[s * 32], 512);       // P chunk -> f16
#pragma unroll
        for (int ht = 0; ht < 4; ++ht) {
            v16h bv = load_bt_f16(&vt[(ht * 16) * 34], 34);
            co[ht] = wmma16(ap, bv, co[ht]);
        }
        __syncthreads();
    }
#pragma unroll
    for (int ht = 0; ht < 4; ++ht)
#pragma unroll
        for (int r = 0; r < 8; ++r) {
            int m = r + 8 * g;
            float val = co[ht][r] * sinv[m];
            o[((size_t)(b * Ss + qt * 16 + m) * DM) + hd * HDim + ht * 16 + ln] =
                (half_t)val;
        }
}

// ---------------- scalar head on last row of each batch --------------------
__global__ __launch_bounds__(256) void head_kernel(
    const float* __restrict__ h, const float* __restrict__ fcW,
    const float* __restrict__ fcb, float* __restrict__ out) {
    int b = blockIdx.x, tid = threadIdx.x;
    const float* row = h + (size_t)(b * Ss + Ss - 1) * DM;
    float s = row[tid] * fcW[tid] + row[tid + 256] * fcW[tid + 256];
#pragma unroll
    for (int off = 16; off; off >>= 1) s += __shfl_xor(s, off, 32);
    __shared__ float red[8];
    if ((tid & 31) == 0) red[tid >> 5] = s;
    __syncthreads();
    if (tid == 0) {
        float t = 0.f;
#pragma unroll
        for (int i = 0; i < 8; ++i) t += red[i];
        out[b] = t + fcb[0];
    }
}

// ---------------- launcher -------------------------------------------------
extern "C" void kernel_launch(void* const* d_in, const int* in_sizes, int n_in,
                              void* d_out, int out_size, void* d_ws,
                              size_t ws_size, hipStream_t stream) {
    (void)in_sizes; (void)n_in; (void)out_size; (void)ws_size;
    const float* x        = (const float*)d_in[0];
    const int*   deg      = (const int*)d_in[1];
    const int*   sp       = (const int*)d_in[2];
    const float* edgeA    = (const float*)d_in[3];
    const int*   lengths  = (const int*)d_in[4];
    const float* nodeW    = (const float*)d_in[5];
    const float* nodeb    = (const float*)d_in[6];
    const float* degtab   = (const float*)d_in[7];
    const float* sptab    = (const float*)d_in[8];
    const float* edgeW    = (const float*)d_in[9];
    const float* eprojW   = (const float*)d_in[10];
    const float* eprojb   = (const float*)d_in[11];
    const float* Wq       = (const float*)d_in[12];
    const float* bq       = (const float*)d_in[13];
    const float* Wk       = (const float*)d_in[14];
    const float* bk       = (const float*)d_in[15];
    const float* Wv       = (const float*)d_in[16];
    const float* bv       = (const float*)d_in[17];
    const float* Wo       = (const float*)d_in[18];
    const float* bo       = (const float*)d_in[19];
    const float* W1       = (const float*)d_in[20];
    const float* b1       = (const float*)d_in[21];
    const float* W2       = (const float*)d_in[22];
    const float* b2       = (const float*)d_in[23];
    const float* ln1s     = (const float*)d_in[24];
    const float* ln1b     = (const float*)d_in[25];
    const float* ln2s     = (const float*)d_in[26];
    const float* ln2b     = (const float*)d_in[27];
    const float* vnode    = (const float*)d_in[28];
    const float* fcW      = (const float*)d_in[29];
    const float* fcb      = (const float*)d_in[30];
    float* out = (float*)d_out;

    // workspace layout (bytes, 1MB-aligned regions)
    char* ws = (char*)d_ws;
    const size_t MT = (size_t)Bb * Ss;                 // 8192 rows
    float*  h_buf  = (float*)(ws + 0);                 // 16 MB
    float*  biasB  = (float*)(ws + (16u << 20));       // 16 MB [B,S,S]
    half_t* xn_h   = (half_t*)(ws + (32u << 20));      // 8 MB each below
    half_t* q_h    = (half_t*)(ws + (40u << 20));
    half_t* k_h    = (half_t*)(ws + (48u << 20));
    half_t* v_h    = (half_t*)(ws + (56u << 20));
    half_t* o_h    = (half_t*)(ws + (64u << 20));
    half_t* mid_h  = (half_t*)(ws + (72u << 20));
    half_t* wT     = (half_t*)(ws + (80u << 20));      // 36 * 512KB = 18 MB
    float*  wcomb  = (float*)(ws + (99u << 20));

    // 1) weights -> f16 transposed
    cvt_weights_kernel<<<dim3(1024, 1, 36), 256, 0, stream>>>(Wq, Wk, Wv, Wo,
                                                              W1, W2, wT);
    // 2) node embedding (+ vnode rows)
    embed_kernel<<<dim3(2, (unsigned)MT), 256, 0, stream>>>(x, deg, nodeW,
                                                            nodeb, degtab,
                                                            vnode, h_buf);
    // 3) fused attention bias
    wcomb_kernel<<<1, 32, 0, stream>>>(edgeW, eprojW, wcomb);
    bias_kernel<<<(Bb * Ss * Ss) / 256, 256, 0, stream>>>(sp, sptab, edgeA,
                                                          wcomb, eprojb,
                                                          lengths, biasB);

    dim3 gGemm(MT / 128, DM / 64);
    dim3 gAttn(Ss / 16, Hh, Bb);
    const size_t DD = (size_t)DM * DM;
    for (int l = 0; l < Ll; ++l) {
        const half_t* wq = wT + (size_t)(l * 6 + 0) * DD;
        const half_t* wk = wT + (size_t)(l * 6 + 1) * DD;
        const half_t* wv = wT + (size_t)(l * 6 + 2) * DD;
        const half_t* wo = wT + (size_t)(l * 6 + 3) * DD;
        const half_t* w1 = wT + (size_t)(l * 6 + 4) * DD;
        const half_t* w2 = wT + (size_t)(l * 6 + 5) * DD;

        ln_kernel<<<(unsigned)MT, 256, 0, stream>>>(h_buf, ln1s + l * DM,
                                                    ln1b + l * DM, xn_h);
        gemm_kernel<0, 0, 0, 1><<<gGemm, 256, 0, stream>>>(
            xn_h, wq, bq + l * DM, nullptr, nullptr, q_h);
        gemm_kernel<0, 0, 0, 1><<<gGemm, 256, 0, stream>>>(
            xn_h, wk, bk + l * DM, nullptr, nullptr, k_h);
        gemm_kernel<0, 0, 0, 1><<<gGemm, 256, 0, stream>>>(
            xn_h, wv, bv + l * DM, nullptr, nullptr, v_h);
        attn_kernel<<<gAttn, 32, 0, stream>>>(q_h, k_h, v_h, biasB, o_h);
        gemm_kernel<0, 1, 1, 0><<<gGemm, 256, 0, stream>>>(
            o_h, wo, bo + l * DM, h_buf, h_buf, nullptr);      // h += o@Wo+bo
        ln_kernel<<<(unsigned)MT, 256, 0, stream>>>(h_buf, ln2s + l * DM,
                                                    ln2b + l * DM, xn_h);
        gemm_kernel<1, 0, 0, 1><<<gGemm, 256, 0, stream>>>(
            xn_h, w1, b1 + l * DM, nullptr, nullptr, mid_h);   // relu
        gemm_kernel<0, 1, 1, 0><<<gGemm, 256, 0, stream>>>(
            mid_h, w2, b2 + l * DM, h_buf, h_buf, nullptr);    // h += ffn
    }
    head_kernel<<<Bb, 256, 0, stream>>>(h_buf, fcW, fcb, out);
}
